// E2ENetwork_78211354460676
// MI455X (gfx1250) — compile-verified, matrix operand order checked by reading, not run on hardware
//
#include <hip/hip_runtime.h>
#include <hip/hip_bf16.h>

// ---------------------------------------------------------------- constants
#define cB 8
#define cN 128
#define cD 512
#define cK 49
#define cC 2000
#define cCp 2048       // padded C for the final GEMM weight
#define cM (cB * cN)   // 1024 rows
#define STEPS 4

// ---------------------------------------------------------------- types
typedef __attribute__((ext_vector_type(16))) __bf16 v16bf;
typedef __attribute__((ext_vector_type(8)))  float  v8f;
typedef __attribute__((ext_vector_type(8)))  unsigned short us8;
typedef __attribute__((ext_vector_type(4)))  unsigned short us4;

union FragU { us8 u[2]; v16bf v; };

__device__ __forceinline__ unsigned short f2bf(float x) {
    union { float f; unsigned int u; } c; c.f = x;
    unsigned int r = c.u + 0x7FFFu + ((c.u >> 16) & 1u);  // round-nearest-even
    return (unsigned short)(r >> 16);
}

// ---------------------------------------------------------------- WMMA GEMM
// All operands bf16, row-major over K:  A[m][k] (lda), Wt[n][k] (ldw).
// C[m,n] = act( sum_k A[m,k]*Wt[n,k] + bias[n] )   (fp32 out)
// Per ISA 10.9, row-major 16-bit fragments load directly with B128 —
// no LDS, no barriers: 8 global_load_b128 + 4 v_wmma per K-step per wave.
#define TM 128
#define TN 64
#define TK 32

__global__ __launch_bounds__(256) void wmma_gemm(
    const unsigned short* __restrict__ A, const unsigned short* __restrict__ Wt,
    const float* __restrict__ bias, float* __restrict__ C,
    int M, int Nn, int Kk, int lda, int ldw, int ldc,
    long long sA, long long sW, long long sC, int act)
{
    const long long z = blockIdx.z;
    A += z * sA;  Wt += z * sW;  C += z * sC;

    const int tid  = threadIdx.x;
    const int lane = tid & 31;
    const int wv   = tid >> 5;
    const int wr   = wv >> 1;       // 0..3 : wave row (32 rows)
    const int wc   = wv & 1;        // 0..1 : wave col (32 cols)
    const int m0   = blockIdx.y * TM;
    const int n0   = blockIdx.x * TN;
    const int l16  = lane & 15;
    const int half = lane >> 4;
    const int kb   = half * 8;      // K-phase per lane-half (elems 0-7 -> kb.., 8-15 -> 16+kb..)

    const unsigned short* a0 = A  + (size_t)(m0 + wr * 32 + l16) * lda + kb;
    const unsigned short* a1 = a0 + (size_t)16 * lda;
    const unsigned short* b0 = Wt + (size_t)(n0 + wc * 32 + l16) * ldw + kb;
    const unsigned short* b1 = b0 + (size_t)16 * ldw;

    v8f acc[2][2] = {};

#pragma unroll 2
    for (int k0 = 0; k0 < Kk; k0 += TK) {
        FragU fa0, fa1, fb0, fb1;
        fa0.u[0] = *(const us8*)(a0 + k0);
        fa0.u[1] = *(const us8*)(a0 + k0 + 16);
        fa1.u[0] = *(const us8*)(a1 + k0);
        fa1.u[1] = *(const us8*)(a1 + k0 + 16);
        fb0.u[0] = *(const us8*)(b0 + k0);
        fb0.u[1] = *(const us8*)(b0 + k0 + 16);
        fb1.u[0] = *(const us8*)(b1 + k0);
        fb1.u[1] = *(const us8*)(b1 + k0 + 16);
        acc[0][0] = __builtin_amdgcn_wmma_f32_16x16x32_bf16(
            false, fa0.v, false, fb0.v, (short)0, acc[0][0], false, false);
        acc[0][1] = __builtin_amdgcn_wmma_f32_16x16x32_bf16(
            false, fa0.v, false, fb1.v, (short)0, acc[0][1], false, false);
        acc[1][0] = __builtin_amdgcn_wmma_f32_16x16x32_bf16(
            false, fa1.v, false, fb0.v, (short)0, acc[1][0], false, false);
        acc[1][1] = __builtin_amdgcn_wmma_f32_16x16x32_bf16(
            false, fa1.v, false, fb1.v, (short)0, acc[1][1], false, false);
    }

    // store: C/D layout — lanes0-15: M=r, N=lane ; lanes16-31: M=8+r
#pragma unroll
    for (int ti = 0; ti < 2; ++ti) {
#pragma unroll
        for (int tj = 0; tj < 2; ++tj) {
            const int col = n0 + wc * 32 + tj * 16 + l16;
            if (col < Nn) {
                const float bsv = bias ? bias[col] : 0.f;
#pragma unroll
                for (int r = 0; r < 8; ++r) {
                    const int row = m0 + wr * 32 + ti * 16 + half * 8 + r;
                    float v = acc[ti][tj][r] + bsv;
                    if (act == 1) v = v > 0.f ? v : (__expf(v) - 1.f);   // ELU
                    C[(size_t)row * ldc + col] = v;
                }
            }
        }
    }
}

// ---------------------------------------------------------------- attention
// One workgroup per (b,n). Stage 100KB knowledge slice in LDS (320KB/WGP),
// logits[k] = slice(:,k) . s  -> softmax(49) -> read[d] = slice(d,:) . attn
__global__ __launch_bounds__(256) void attn_kernel(
    const float* __restrict__ knowledge, const float* __restrict__ mcoef,
    const float* __restrict__ u1, const float* __restrict__ u2,
    const float* __restrict__ cscal, float* __restrict__ readout)
{
    extern __shared__ float sm[];
    float* sl = sm;                 // cD*cK slice
    float* sv = sm + cD * cK;       // s[d]
    float* lg = sv + cD;            // logits / attn

    const int r = blockIdx.x;       // flat (b,n)
    const float* base = knowledge + (size_t)r * cD * cK;

    __builtin_prefetch(base + cD * cK, 0, 1);

    const float4* b4  = (const float4*)base;
    float4*       sl4 = (float4*)sl;
    for (int i = threadIdx.x; i < (cD * cK) / 4; i += 256) sl4[i] = b4[i];

    for (int d = threadIdx.x; d < cD; d += 256) {
        size_t idx = (size_t)r * cD + d;
        sv[d] = mcoef[idx] * u1[idx] + u2[idx];
    }
    __syncthreads();

    if (threadIdx.x < cK) {
        int k = threadIdx.x;
        float acc = 0.f;
        for (int d = 0; d < cD; ++d) acc += sl[d * cK + k] * sv[d];
        lg[k] = acc + cscal[r];
    }
    __syncthreads();
    if (threadIdx.x == 0) {
        float mx = lg[0];
        for (int k = 1; k < cK; ++k) mx = fmaxf(mx, lg[k]);
        float s = 0.f;
        for (int k = 0; k < cK; ++k) { float e = __expf(lg[k] - mx); lg[k] = e; s += e; }
        float inv = 1.f / s;
        for (int k = 0; k < cK; ++k) lg[k] *= inv;
    }
    __syncthreads();
    for (int d = threadIdx.x; d < cD; d += 256) {
        float acc = 0.f;
#pragma unroll
        for (int k = 0; k < cK; ++k) acc += lg[k] * sl[d * cK + k];
        readout[(size_t)r * cD + d] = acc;
    }
}

// ---------------------------------------------------------------- data-movement kernels
// straight fp32 -> bf16 convert (vectorized), total must be multiple of 4
__global__ void conv_kernel(unsigned short* __restrict__ dst,
                            const float* __restrict__ src, size_t total4)
{
    size_t i = (size_t)blockIdx.x * 256 + threadIdx.x;
    if (i < total4) {
        float4 f = ((const float4*)src)[i];
        us4 p; p.x = f2bf(f.x); p.y = f2bf(f.y); p.z = f2bf(f.z); p.w = f2bf(f.w);
        ((us4*)dst)[i] = p;
    }
}

// dst[z][n][k] = f2bf(src[z][k*N + n])  for n<N else 0 ; dst is [Np][Kk] row-major
__global__ void tconv_kernel(unsigned short* __restrict__ dst,
                             const float* __restrict__ src,
                             int Kk, int N, int Np, long long sSrc, long long sDst)
{
    const long long z = blockIdx.y;
    size_t i = (size_t)blockIdx.x * 256 + threadIdx.x;
    if (i < (size_t)Np * Kk) {
        int n = (int)(i / Kk), k = (int)(i % Kk);
        unsigned short v = 0;
        if (n < N) v = f2bf(src[z * sSrc + (size_t)k * N + n]);
        dst[z * sDst + i] = v;
    }
}

// bf16 concat pack: dst row r = [s0 | s1 | s2*mul2]
__global__ void pack3_bf16_kernel(unsigned short* __restrict__ dst,
                                  const float* __restrict__ s0, int k0,
                                  const float* __restrict__ s1, int k1,
                                  const float* __restrict__ s2,
                                  const float* __restrict__ mul2, int k2, int M)
{
    const int Wd = k0 + k1 + k2;
    const size_t total = (size_t)M * Wd;
    for (size_t i = (size_t)blockIdx.x * 256 + threadIdx.x; i < total;
         i += (size_t)gridDim.x * 256) {
        int r = (int)(i / Wd), c = (int)(i % Wd);
        float v;
        if (c < k0) v = s0[(size_t)r * k0 + c];
        else if (c < k0 + k1) v = s1[(size_t)r * k1 + (c - k0)];
        else {
            int cc = c - k0 - k1;
            v = s2[(size_t)r * k2 + cc];
            if (mul2) v *= mul2[(size_t)r * k2 + cc];
        }
        dst[i] = f2bf(v);
    }
}

__global__ void bcast_mem_kernel(float* __restrict__ mem,
                                 const float* __restrict__ mem0,
                                 int nd, size_t total)
{
    size_t i = (size_t)blockIdx.x * 256 + threadIdx.x;
    if (i < total) mem[i] = mem0[i % nd];
}

__global__ void vmul_kernel(float* __restrict__ v, const float* __restrict__ q,
                            const float* __restrict__ wattn, int Dd, size_t total)
{
    size_t i = (size_t)blockIdx.x * 256 + threadIdx.x;
    if (i < total) v[i] = q[i] * wattn[i % Dd];
}

__global__ void cscal_kernel(float* __restrict__ cs, const float* __restrict__ v,
                             const float* __restrict__ brcat,
                             const float* __restrict__ battn, int Dd, int M)
{
    int r = blockIdx.x * 256 + threadIdx.x;
    if (r < M) {
        float acc = 0.f;
        for (int f = 0; f < Dd; ++f) acc += brcat[f] * v[(size_t)r * Dd + f];
        cs[r] = acc + battn[0];
    }
}

// ---------------------------------------------------------------- host helpers
static void gemm(hipStream_t st, const unsigned short* A, const unsigned short* Wt,
                 const float* bias, float* C, int M, int N, int K,
                 int lda, int ldw, int ldc,
                 long long sA, long long sW, long long sC, int batch, int act)
{
    dim3 g((N + TN - 1) / TN, M / TM, batch), b(256);
    hipLaunchKernelGGL(wmma_gemm, g, b, 0, st, A, Wt, bias, C, M, N, K,
                       lda, ldw, ldc, sA, sW, sC, act);
}

static void conv(hipStream_t st, unsigned short* dst, const float* src, size_t total)
{
    size_t t4 = total / 4;
    hipLaunchKernelGGL(conv_kernel, dim3((t4 + 255) / 256), dim3(256), 0, st,
                       dst, src, t4);
}

static void tconv(hipStream_t st, unsigned short* dst, const float* src,
                  int K, int N, int Np, long long sSrc, long long sDst, int batch)
{
    size_t tot = (size_t)Np * K;
    dim3 g((unsigned)((tot + 255) / 256), batch);
    hipLaunchKernelGGL(tconv_kernel, g, dim3(256), 0, st, dst, src, K, N, Np,
                       sSrc, sDst);
}

extern "C" void kernel_launch(void* const* d_in, const int* in_sizes, int n_in,
                              void* d_out, int out_size, void* d_ws, size_t ws_size,
                              hipStream_t stream)
{
    const float* question  = (const float*)d_in[0];
    const float* knowledge = (const float*)d_in[1];
    const float* Amat      = (const float*)d_in[2];
    const float* mem0      = (const float*)d_in[3];
    const float* W_in   = (const float*)d_in[4];
    const float* b_in   = (const float*)d_in[5];
    const float* W_out  = (const float*)d_in[6];
    const float* b_out  = (const float*)d_in[7];
    const float* W_rmem = (const float*)d_in[8];
    const float* b_rmem = (const float*)d_in[9];
    const float* W_rcat = (const float*)d_in[10];
    const float* b_rcat = (const float*)d_in[11];
    const float* W_attn = (const float*)d_in[12];
    const float* b_attn = (const float*)d_in[13];
    const float* W_tr   = (const float*)d_in[14];
    const float* b_tr   = (const float*)d_in[15];
    const float* W_c1   = (const float*)d_in[16];
    const float* b_c1   = (const float*)d_in[17];
    const float* W_c2   = (const float*)d_in[18];
    const float* b_c2   = (const float*)d_in[19];

    float* ws = (float*)d_ws;
    const size_t MD = (size_t)cM * cD;            // 524288

    // fp32 region
    float* mem   = ws;
    float* ins   = ws + 1 * MD;
    float* outs  = ws + 2 * MD;
    float* a_in  = ws + 3 * MD;
    float* a_out = ws + 4 * MD;
    float* mcoef = ws + 5 * MD;
    float* readb = ws + 6 * MD;
    float* u1    = ws + 7 * MD;
    float* u2    = ws + 8 * MD;
    float* vbuf  = ws + 9 * MD;
    float* hbuf  = ws + 10 * MD;
    float* cscal = ws + 11 * MD;                  // cM floats

    // bf16 region (ushort), after 12*MD floats
    unsigned short* bb = (unsigned short*)(ws + 12 * MD);
    size_t o = 0;
    auto take = [&](size_t n) { unsigned short* p = bb + o; o += n; return p; };
    unsigned short* memb    = take(MD);                       // mem bf16 [m][k]
    unsigned short* vbufb   = take(MD);                       // v bf16
    unsigned short* hbufb   = take(MD);                       // h bf16
    unsigned short* Amatb   = take((size_t)cB * cN * cN);     // A row-major
    unsigned short* insb_t  = take((size_t)cB * cD * cN);     // [b][d][j]
    unsigned short* outsb_t = take((size_t)cB * cD * cN);
    unsigned short* acatb   = take(3 * MD);                   // [m][3D]
    unsigned short* jcatb   = take(3 * MD);
    unsigned short* hcatb   = take(2 * MD);                   // [m][2D]
    unsigned short* Wrcatb  = take((size_t)2 * cD * cD);      // straight convert
    unsigned short* Wint    = take((size_t)cD * cD);          // [n][k]
    unsigned short* Woutt   = take((size_t)cD * cD);
    unsigned short* Wrmemt  = take((size_t)cD * 3 * cD);      // [512][1536]
    unsigned short* Wtrt    = take((size_t)cD * 3 * cD);
    unsigned short* Wc1t    = take((size_t)cD * 2 * cD);      // [512][1024]
    unsigned short* Wc2t    = take((size_t)cCp * cD);         // [2048][512], zero-padded

    // ---- init
    hipLaunchKernelGGL(bcast_mem_kernel, dim3((MD + 255) / 256), dim3(256), 0, stream,
                       mem, mem0, cN * cD, MD);
    hipLaunchKernelGGL(vmul_kernel, dim3((MD + 255) / 256), dim3(256), 0, stream,
                       vbuf, question, W_attn, cD, MD);
    conv(stream, memb,  mem,    MD);
    conv(stream, vbufb, vbuf,   MD);
    conv(stream, Wrcatb, W_rcat, (size_t)2 * cD * cD);        // [d][f] both halves
    conv(stream, Amatb, Amat,   (size_t)cB * cN * cN);
    // static weight transposes: dst[n][k] = W[k][n]
    tconv(stream, Wint,   W_in,   cD,     cD, cD,  0, 0, 1);
    tconv(stream, Woutt,  W_out,  cD,     cD, cD,  0, 0, 1);
    tconv(stream, Wrmemt, W_rmem, 3 * cD, cD, cD,  0, 0, 1);
    tconv(stream, Wtrt,   W_tr,   3 * cD, cD, cD,  0, 0, 1);
    tconv(stream, Wc1t,   W_c1,   2 * cD, cD, cD,  0, 0, 1);
    tconv(stream, Wc2t,   W_c2,   cD,     cC, cCp, 0, 0, 1);

    // u1 = v @ W1^T, u2 = v @ W2^T  (Wrcatb rows d, cols f: already [n][k])
    gemm(stream, vbufb, Wrcatb, nullptr, u1, cM, cD, cD, cD, cD, cD, 0, 0, 0, 1, 0);
    gemm(stream, vbufb, Wrcatb + (size_t)cD * cD, nullptr, u2,
         cM, cD, cD, cD, cD, cD, 0, 0, 0, 1, 0);
    hipLaunchKernelGGL(cscal_kernel, dim3((cM + 255) / 256), dim3(256), 0, stream,
                       cscal, vbuf, b_rcat, b_attn, cD, cM);

    const int attn_shmem = (cD * cK + cD + 64) * (int)sizeof(float);
    const size_t tot3 = (size_t)cM * 3 * cD;
    const size_t tot2 = (size_t)cM * 2 * cD;

    for (int step = 0; step < STEPS; ++step) {
        // ins/outs = mem @ W_{in,out} + b
        gemm(stream, memb, Wint,  b_in,  ins,  cM, cD, cD, cD, cD, cD, 0, 0, 0, 1, 0);
        gemm(stream, memb, Woutt, b_out, outs, cM, cD, cD, cD, cD, cD, 0, 0, 0, 1, 0);
        // transpose-convert per batch: insb_t[b][d][j] = bf16(ins[b][j][d])
        tconv(stream, insb_t,  ins,  cN, cD, cD,
              (long long)cN * cD, (long long)cD * cN, cB);
        tconv(stream, outsb_t, outs, cN, cD, cD,
              (long long)cN * cD, (long long)cD * cN, cB);
        // a_in[b] = A[b] @ ins[b]
        gemm(stream, Amatb, insb_t,  nullptr, a_in,  cN, cD, cN, cN, cN, cD,
             (long long)cN * cN, (long long)cD * cN, (long long)cN * cD, cB, 0);
        gemm(stream, Amatb, outsb_t, nullptr, a_out, cN, cD, cN, cN, cN, cD,
             (long long)cN * cN, (long long)cD * cN, (long long)cN * cD, cB, 0);
        // acat = [a_in | a_out | mem] (bf16) ; m = acat @ W_rmem + b_rmem
        hipLaunchKernelGGL(pack3_bf16_kernel, dim3((tot3 + 255) / 256), dim3(256),
                           0, stream, acatb, a_in, cD, a_out, cD, mem,
                           (const float*)nullptr, cD, cM);
        gemm(stream, acatb, Wrmemt, b_rmem, mcoef, cM, cD, 3 * cD, 3 * cD, 3 * cD, cD,
             0, 0, 0, 1, 0);
        // attention
        hipLaunchKernelGGL(attn_kernel, dim3(cM), dim3(256), attn_shmem, stream,
                           knowledge, mcoef, u1, u2, cscal, readb);
        // jcat = [a_in | a_out | read*mem] (bf16) ; mem = elu(jcat @ W_tr + b_tr)
        hipLaunchKernelGGL(pack3_bf16_kernel, dim3((tot3 + 255) / 256), dim3(256),
                           0, stream, jcatb, a_in, cD, a_out, cD, readb, mem, cD, cM);
        gemm(stream, jcatb, Wtrt, b_tr, mem, cM, cD, 3 * cD, 3 * cD, 3 * cD, cD,
             0, 0, 0, 1, /*elu*/1);
        conv(stream, memb, mem, MD);
    }

    // h = elu([mem | question] @ W_c1 + b_c1) ; out = h @ W_c2 + b_c2
    hipLaunchKernelGGL(pack3_bf16_kernel, dim3((tot2 + 255) / 256), dim3(256),
                       0, stream, hcatb, mem, cD, question, cD,
                       (const float*)nullptr, (const float*)nullptr, 0, cM);
    gemm(stream, hcatb, Wc1t, b_c1, hbuf, cM, cD, 2 * cD, 2 * cD, 2 * cD, cD,
         0, 0, 0, 1, 1);
    conv(stream, hbufb, hbuf, MD);
    gemm(stream, hbufb, Wc2t, b_c2, (float*)d_out, cM, cC, cD, cD, cD, cC,
         0, 0, 0, 1, 0);
    (void)in_sizes; (void)n_in; (void)out_size; (void)ws_size;
}
